// DetectionLoss_53240414601535
// MI455X (gfx1250) — compile-verified, matrix operand order checked by reading, not run on hardware
//
#include <hip/hip_runtime.h>
#include <hip/hip_bf16.h>

// ---------------- problem constants (match reference) ----------------
constexpr int   kB = 16;
constexpr int   kC = 21;
constexpr int   kH = 256, kW = 256;
constexpr int   kN = kH * kW;          // 65536 anchors
constexpr int   kM = 20;               // GT boxes per image
constexpr float kPosIou = 0.5f;
constexpr float kNegIou = 0.4f;
constexpr float kAlpha  = 0.25f;
constexpr int   kMinPosTopK = 10;
constexpr int   kNegPosRatio = 3;
constexpr int   kBlk = 256;            // threads per block (8 waves of 32)
constexpr int   kNB4 = 32;             // partial-sum blocks per image in K4

typedef __attribute__((ext_vector_type(16))) _Float16 v16h;
typedef __attribute__((ext_vector_type(8)))  float    v8f;

__device__ __forceinline__ unsigned fkey(float f) {
    unsigned u = __float_as_uint(f);
    return (u & 0x80000000u) ? ~u : (u | 0x80000000u);   // order-preserving map
}

// =====================================================================
// K1: fused per-anchor pass.
//   - decode boxes (b128 anchor loads)
//   - IoU vs 20 GT (LDS-staged), max/argmax; matched box re-fetched with
//     ONE dynamically indexed ds_load_b128 (no 20-way cndmask chain)
//   - focal CE via log-softmax; denominator via v_wmma_f32_16x16x32_f16
//     (f16 exp tile x ones); target/background exp read back from the
//     same LDS tile so the logit register array dies early
//   - writes float4 {max_iou, focal_pos, focal_neg, 1-giou} (b128 store)
//   - integer-atomic counts of strict positives / all negatives
// =====================================================================
__global__ __launch_bounds__(kBlk) void k1_fused(
    const float* __restrict__ cls,      // [B,C,N]
    const float* __restrict__ reg,      // [B,4,N]
    const float* __restrict__ anchors,  // [N,4]
    const float* __restrict__ tboxes,   // [B,M,4]
    const int*   __restrict__ tlabels,  // [B,M]
    float4*      __restrict__ pa,       // [B,N] intermediate
    int*         __restrict__ cnt_pos,  // [B]
    int*         __restrict__ cnt_neg)  // [B]
{
    const int tid  = threadIdx.x;
    const int b    = blockIdx.y;
    const int n    = blockIdx.x * kBlk + tid;
    const int lane = tid & 31;
    const int wave = tid >> 5;

    __shared__ __align__(16) float gtb[kM * 4];
    __shared__ int   gtl[kM];
    __shared__ unsigned expt[kBlk][16];   // 256 anchors x 32 classes (f16 pairs)
    __shared__ float sums[kBlk];
    __shared__ int   s_cp, s_cn;

    if (tid < kM * 4) gtb[tid] = tboxes[b * kM * 4 + tid];
    if (tid < kM)     gtl[tid] = tlabels[b * kM + tid];
    if (tid == 0) { s_cp = 0; s_cn = 0; }
    __syncthreads();

    // ---- decode ----
    const float4 a = ((const float4*)anchors)[n];
    const float aw = a.z - a.x, ah = a.w - a.y;
    const float ax = a.x + 0.5f * aw, ay = a.y + 0.5f * ah;
    const size_t rb = (size_t)b * 4 * kN + n;
    const float r0 = reg[rb], r1 = reg[rb + kN], r2 = reg[rb + 2 * kN], r3 = reg[rb + 3 * kN];
    const float cx = ax + r0 * aw, cy = ay + r1 * ah;
    const float dw = aw * __expf(r2), dh = ah * __expf(r3);
    const float dx1 = cx - 0.5f * dw, dy1 = cy - 0.5f * dh;
    const float dx2 = cx + 0.5f * dw, dy2 = cy + 0.5f * dh;
    const float area_d = dw * dh;

    // ---- IoU vs 20 GT (static unrolled LDS reads -> registers) ----
    float best = -1.0f; int bestm = 0;
    #pragma unroll 4
    for (int m = 0; m < kM; ++m) {
        const float bx1 = gtb[4*m+0], by1 = gtb[4*m+1], bx2 = gtb[4*m+2], by2 = gtb[4*m+3];
        const float iw = fminf(dx2, bx2) - fmaxf(dx1, bx1);
        const float ih = fminf(dy2, by2) - fmaxf(dy1, by1);
        const float inter = fmaxf(iw, 0.f) * fmaxf(ih, 0.f);
        const float uni = area_d + (bx2 - bx1) * (by2 - by1) - inter;
        const float iou = inter / fmaxf(uni, 1e-7f);
        if (iou > best) { best = iou; bestm = m; }
    }

    // ---- GIoU with matched GT: one dynamic ds_load_b128 ----
    float regterm;
    {
        const float4 g = *(const float4*)&gtb[4 * bestm];
        const float iw = fmaxf(fminf(dx2, g.z) - fmaxf(dx1, g.x), 0.f);
        const float ih = fmaxf(fminf(dy2, g.w) - fmaxf(dy1, g.y), 0.f);
        const float inter = iw * ih;
        const float uni = area_d + (g.z - g.x) * (g.w - g.y) - inter;
        const float iou = inter / fmaxf(uni, 1e-7f);
        const float ew = fmaxf(dx2, g.z) - fminf(dx1, g.x);
        const float eh = fmaxf(dy2, g.w) - fminf(dy1, g.y);
        const float enc = fmaxf(ew, 0.f) * fmaxf(eh, 0.f);
        regterm = 1.0f - (iou - (enc - uni) / fmaxf(enc, 1e-7f));
    }
    const int tgt = gtl[bestm];

    // ---- logits: single NT pass (read-once 88MB stream, keep L2 for pa) ----
    float lg[kC];
    float mx = -INFINITY;
    const float* cb = cls + (size_t)b * kC * kN + n;
    #pragma unroll
    for (int c = 0; c < kC; ++c) {
        if (c + 1 < kC) __builtin_prefetch(cb + (size_t)(c + 1) * kN, 0, 0);
        lg[c] = __builtin_nontemporal_load(cb + (size_t)c * kN);
        mx = fmaxf(mx, lg[c]);
    }

    // stage exp tile into LDS (f16, classes 21..31 zero-padded)
    #pragma unroll
    for (int i = 0; i < 16; ++i) {
        union { _Float16 h[2]; unsigned u; } p;
        const int c0 = 2 * i, c1 = 2 * i + 1;
        p.h[0] = (c0 < kC) ? (_Float16)__expf(lg[c0] - mx) : (_Float16)0.0f;
        p.h[1] = (c1 < kC) ? (_Float16)__expf(lg[c1] - mx) : (_Float16)0.0f;
        expt[tid][i] = p.u;
    }

    // read back own-row exp(tgt), exp(0) -- no cross-thread race, no chain
    float e_tgt, e_bg;
    {
        union { unsigned u; _Float16 h[2]; } p;
        p.u = expt[tid][tgt >> 1];
        e_tgt = fmaxf((float)p.h[tgt & 1], 1e-12f);
        p.u = expt[tid][0];
        e_bg  = fmaxf((float)p.h[0], 1e-12f);
    }
    __syncthreads();

    // ---- WMMA row-sum GEMV: D = exp_tile(16x32) x ones(32x16) ----
    v16h ones;
    #pragma unroll
    for (int i = 0; i < 16; ++i) ones[i] = (_Float16)1.0f;

    #pragma unroll
    for (int t = 0; t < 2; ++t) {
        const int tilebase = wave * 32 + t * 16;
        const int row = tilebase + (lane & 15);
        const int half_sel = (lane < 16) ? 0 : 1;  // K chunks per ISA A-layout
        union { unsigned u[8]; v16h v; } av;
        #pragma unroll
        for (int q = 0; q < 4; ++q) {
            av.u[q]     = expt[row][half_sel * 4 + q];      // K 0..7  / 8..15
            av.u[4 + q] = expt[row][8 + half_sel * 4 + q];  // K 16..23 / 24..31
        }
        v8f acc = {};
        acc = __builtin_amdgcn_wmma_f32_16x16x32_f16(
            false, av.v, false, ones, (short)0, acc, false, false);
        if ((lane & 15) == 0) {                 // column n==0 lives in lanes 0,16
            const int mbase = tilebase + ((lane < 16) ? 0 : 8);
            #pragma unroll
            for (int r = 0; r < 8; ++r) sums[mbase + r] = acc[r];
        }
    }
    __syncthreads();

    const float Z = sums[tid];
    const float logZ = __logf(Z);
    const float ce_pos = logZ - __logf(e_tgt);     // mx cancels
    const float ce_neg = logZ - __logf(e_bg);
    const float ptp = e_tgt / Z, ptn = e_bg / Z;   // = exp(-ce)
    const float fp = kAlpha * (1.f - ptp) * (1.f - ptp) * ce_pos;
    const float fn = kAlpha * (1.f - ptn) * (1.f - ptn) * ce_neg;

    pa[(size_t)b * kN + n] = make_float4(best, fp, fn, regterm);

    if (best >= kPosIou) atomicAdd(&s_cp, 1);
    if (best <  kNegIou) atomicAdd(&s_cn, 1);
    __syncthreads();
    if (tid == 0) {
        atomicAdd(&cnt_pos[b], s_cp);
        atomicAdd(&cnt_neg[b], s_cn);
    }
}

// =====================================================================
// K2: per-image positive threshold (10th-largest max-IoU radix select
// when strict count < 10, else key(0.5)), then num_pos recount.
// =====================================================================
__global__ __launch_bounds__(256) void k2_pos_select(
    const float4* __restrict__ pa, const int* __restrict__ cnt_pos,
    unsigned* __restrict__ Tpos, int* __restrict__ num_pos)
{
    const int b = blockIdx.x, tid = threadIdx.x;
    const float4* p = pa + (size_t)b * kN;
    __shared__ int hist[256];
    __shared__ unsigned s_prefix;
    __shared__ int s_k, s_cnt;

    unsigned T;
    if (cnt_pos[b] >= kMinPosTopK) {
        T = fkey(kPosIou);
    } else {
        if (tid == 0) { s_prefix = 0; s_k = kMinPosTopK; }
        __syncthreads();
        unsigned pmask = 0;
        for (int shift = 24; shift >= 0; shift -= 8) {
            hist[tid] = 0;
            __syncthreads();
            const unsigned prefix = s_prefix;
            for (int i = tid; i < kN; i += 256) {
                const unsigned key = fkey(p[i].x);
                if ((key & pmask) == prefix) atomicAdd(&hist[(key >> shift) & 255], 1);
            }
            __syncthreads();
            if (tid == 0) {
                int k = s_k, cum = 0, digit = 0;
                for (int bin = 255; bin >= 0; --bin) {
                    cum += hist[bin];
                    if (cum >= k) { digit = bin; s_k = k - (cum - hist[bin]); break; }
                }
                s_prefix = prefix | ((unsigned)digit << shift);
            }
            __syncthreads();
            pmask |= (255u << shift);
        }
        T = s_prefix;
    }
    if (tid == 0) s_cnt = 0;
    __syncthreads();
    int local = 0;
    for (int i = tid; i < kN; i += 256) if (fkey(p[i].x) >= T) local++;
    atomicAdd(&s_cnt, local);
    __syncthreads();
    if (tid == 0) { Tpos[b] = T; num_pos[b] = s_cnt; }
}

// =====================================================================
// K3: per-image negative threshold = K-th smallest score among negatives
// (K = 3*num_pos); +inf sentinel key for non-negatives; skip if no
// subsampling needed.
// =====================================================================
__global__ __launch_bounds__(256) void k3_neg_select(
    const float4* __restrict__ pa, const float* __restrict__ scores,
    const int* __restrict__ cnt_neg, const int* __restrict__ num_pos,
    unsigned* __restrict__ Tneg)
{
    const int b = blockIdx.x, tid = threadIdx.x;
    const int K = kNegPosRatio * num_pos[b];
    if (cnt_neg[b] <= K) { if (tid == 0) Tneg[b] = 0xFFFFFFFFu; return; }
    if (K <= 0)          { if (tid == 0) Tneg[b] = 0u;          return; }

    const float4* p = pa + (size_t)b * kN;
    const float*  s = scores + (size_t)b * kN;
    __shared__ int hist[256];
    __shared__ unsigned s_prefix;
    __shared__ int s_k;
    if (tid == 0) { s_prefix = 0; s_k = K; }
    __syncthreads();
    unsigned pmask = 0;
    for (int shift = 24; shift >= 0; shift -= 8) {
        hist[tid] = 0;
        __syncthreads();
        const unsigned prefix = s_prefix;
        for (int i = tid; i < kN; i += 256) {
            const unsigned key = (p[i].x < kNegIou) ? fkey(s[i]) : 0xFFFFFFFFu;
            if ((key & pmask) == prefix) atomicAdd(&hist[(key >> shift) & 255], 1);
        }
        __syncthreads();
        if (tid == 0) {
            int k = s_k, cum = 0, digit = 255;
            for (int bin = 0; bin < 256; ++bin) {
                cum += hist[bin];
                if (cum >= k) { digit = bin; s_k = k - (cum - hist[bin]); break; }
            }
            s_prefix = prefix | ((unsigned)digit << shift);
        }
        __syncthreads();
        pmask |= (255u << shift);
    }
    if (tid == 0) Tneg[b] = s_prefix;
}

// =====================================================================
// K4: masked partial sums per (image, block of 2048 anchors).
// =====================================================================
__global__ __launch_bounds__(256) void k4_partials(
    const float4* __restrict__ pa, const float* __restrict__ scores,
    const unsigned* __restrict__ Tpos, const unsigned* __restrict__ Tneg,
    float* __restrict__ partials)
{
    const int b = blockIdx.x, blk = blockIdx.y, tid = threadIdx.x;
    const int base = blk * (kN / kNB4);
    const unsigned tp = Tpos[b], tn = Tneg[b];
    const float4* p = pa + (size_t)b * kN;
    const float*  s = scores + (size_t)b * kN;
    __builtin_prefetch(s + base + tid, 0, 0);

    float cp = 0.f, cn = 0.f, rg = 0.f, np = 0.f, nn = 0.f;
    #pragma unroll
    for (int j = 0; j < kN / kNB4 / 256; ++j) {
        const int n = base + j * 256 + tid;
        const float4 v = p[n];
        const bool pos = fkey(v.x) >= tp;
        const bool neg = (v.x < kNegIou) && (fkey(s[n]) <= tn);
        if (pos) { cp += v.y; rg += v.w; np += 1.f; }
        if (neg) { cn += v.z; nn += 1.f; }
    }
    __shared__ float sm[5][256];
    sm[0][tid] = cp; sm[1][tid] = cn; sm[2][tid] = rg; sm[3][tid] = np; sm[4][tid] = nn;
    __syncthreads();
    for (int st = 128; st > 0; st >>= 1) {
        if (tid < st)
            #pragma unroll
            for (int q = 0; q < 5; ++q) sm[q][tid] += sm[q][tid + st];
        __syncthreads();
    }
    if (tid == 0) {
        float* o = partials + ((size_t)b * kNB4 + blk) * 8;
        #pragma unroll
        for (int q = 0; q < 5; ++q) o[q] = sm[q][0];
    }
}

// =====================================================================
// K5: per-image loss from partials (deterministic tree, no float atomics).
// =====================================================================
__global__ __launch_bounds__(32) void k5_image_loss(
    const float* __restrict__ partials, float* __restrict__ img_loss)
{
    const int b = blockIdx.x, tid = threadIdx.x;
    __shared__ float sm[5][32];
    const float* o = partials + ((size_t)b * kNB4 + tid) * 8;
    #pragma unroll
    for (int q = 0; q < 5; ++q) sm[q][tid] = o[q];
    __syncthreads();
    if (tid == 0) {
        float acc[5] = {0.f, 0.f, 0.f, 0.f, 0.f};
        for (int j = 0; j < kNB4; ++j)
            for (int q = 0; q < 5; ++q) acc[q] += sm[q][j];
        const float np = acc[3], nn = acc[4];
        const float cls_loss = (acc[0] + acc[1]) / fmaxf(np + nn, 1.f);
        const float reg_loss = acc[2] / fmaxf(np, 1.f);
        img_loss[b] = cls_loss + reg_loss;   // REG_LOSS_WEIGHT = 1
    }
}

// K6: mean over images -> scalar output.
__global__ __launch_bounds__(32) void k6_final(
    const float* __restrict__ img_loss, float* __restrict__ out)
{
    if (threadIdx.x == 0) {
        float s = 0.f;
        for (int b = 0; b < kB; ++b) s += img_loss[b];
        out[0] = s / (float)kB;
    }
}

// =====================================================================
extern "C" void kernel_launch(void* const* d_in, const int* in_sizes, int n_in,
                              void* d_out, int out_size, void* d_ws, size_t ws_size,
                              hipStream_t stream) {
    (void)in_sizes; (void)n_in; (void)out_size; (void)ws_size;
    const float* cls     = (const float*)d_in[0];
    const float* reg     = (const float*)d_in[1];
    const float* anchors = (const float*)d_in[2];
    const float* tboxes  = (const float*)d_in[3];
    const int*   tlabels = (const int*)d_in[4];
    const float* scores  = (const float*)d_in[5];

    char* ws = (char*)d_ws;
    const size_t PA_BYTES = (size_t)kB * kN * sizeof(float4);   // 16.78 MB
    float4*   pa       = (float4*)ws;
    int*      cnt_pos  = (int*)(ws + PA_BYTES);
    int*      cnt_neg  = cnt_pos + kB;
    unsigned* Tpos     = (unsigned*)(cnt_neg + kB);
    int*      num_pos  = (int*)(Tpos + kB);
    unsigned* Tneg     = (unsigned*)(num_pos + kB);
    float*    partials = (float*)(Tneg + kB);                   // [B][32][8]
    float*    img_loss = partials + (size_t)kB * kNB4 * 8;

    hipMemsetAsync(cnt_pos, 0, 2 * kB * sizeof(int), stream);

    dim3 g1(kN / kBlk, kB);
    k1_fused<<<g1, kBlk, 0, stream>>>(cls, reg, anchors, tboxes, tlabels,
                                      pa, cnt_pos, cnt_neg);
    k2_pos_select<<<kB, 256, 0, stream>>>(pa, cnt_pos, Tpos, num_pos);
    k3_neg_select<<<kB, 256, 0, stream>>>(pa, scores, cnt_neg, num_pos, Tneg);
    dim3 g4(kB, kNB4);
    k4_partials<<<g4, 256, 0, stream>>>(pa, scores, Tpos, Tneg, partials);
    k5_image_loss<<<kB, 32, 0, stream>>>(partials, img_loss);
    k6_final<<<1, 32, 0, stream>>>(img_loss, (float*)d_out);
}
